// DiscreteTimeS4_31387620999386
// MI455X (gfx1250) — matrix operand order repaired and checked by reading, not compiled, not run on hardware
//
#include <hip/hip_runtime.h>

typedef __attribute__((ext_vector_type(16))) __bf16    v16bf;
typedef __attribute__((ext_vector_type(8)))  float     v8f;
typedef __attribute__((ext_vector_type(4)))  unsigned  v4u;
typedef __attribute__((ext_vector_type(8)))  int       v8i;
typedef __attribute__((ext_vector_type(4)))  int       v4i;

#define BSZ   8
#define TLEN  4096
#define D_IN  512
#define PDIM  256
#define HDIM  256
#define D_OUT 512
#define MROWS (BSZ * TLEN)      // 32768

// chunked-scan parameters
#define NC      16
#define LCHUNK  (TLEN / NC)     // 256

#if defined(__HIP_DEVICE_COMPILE__) && __has_builtin(__builtin_amdgcn_tensor_load_to_lds)
#define USE_TDM 1
#else
#define USE_TDM 0
#endif

template <class A, class B> struct is_same_t { static constexpr bool value = false; };
template <class A> struct is_same_t<A, A>    { static constexpr bool value = true;  };

// ---------------------------------------------------------------------------
// TDM: async DMA of a 128(rows) x 32(K) bf16 tile, K-contiguous, into LDS.
// Tensor is [rows, k_total] bf16 row-major; gptr points at the tile origin.
// D# layout per CDNA5 ISA 8.3/8.4 (group0 + group1, 2-D tile, no pad/iter).
// ---------------------------------------------------------------------------
#if USE_TDM
__device__ __forceinline__ void tdm_load_tile_128x32(unsigned lds_addr,
                                                     const void* gptr,
                                                     unsigned k_total,
                                                     unsigned rows) {
  unsigned long long ga = (unsigned long long)(size_t)gptr;
  v4u g0;
  g0[0] = 1u;                                            // count=1, user desc
  g0[1] = lds_addr;                                      // LDS byte address
  g0[2] = (unsigned)(ga & 0xFFFFFFFFu);                  // global_addr[31:0]
  g0[3] = (unsigned)((ga >> 32) & 0x1FFFFFFu) | (2u << 30); // [56:32] | type=2
  v8i g1;
  g1[0] = (int)(1u << 16);                               // data_size=1 (2B), mask=0
  g1[1] = (int)((k_total & 0xFFFFu) << 16);              // tensor_dim0[15:0]
  g1[2] = (int)(((k_total >> 16) & 0xFFFFu) |            // tensor_dim0[31:16]
                ((rows & 0xFFFFu) << 16));               // tensor_dim1[15:0]
  g1[3] = (int)(((rows >> 16) & 0xFFFFu) |               // tensor_dim1[31:16]
                (32u << 16));                            // tile_dim0 = 32
  g1[4] = (int)128u;                                     // tile_dim1=128, tile_dim2=0
  g1[5] = (int)k_total;                                  // tensor_dim0_stride[31:0]
  g1[6] = 0;                                             // stride0_hi | stride1_lo
  g1[7] = 0;
  v4i z4 = {0, 0, 0, 0};
#if defined(__clang_major__) && __clang_major__ >= 23
  v8i z8 = {0, 0, 0, 0, 0, 0, 0, 0};
  __builtin_amdgcn_tensor_load_to_lds(g0, g1, z4, z4, z8, 0);
#else
  __builtin_amdgcn_tensor_load_to_lds(g0, g1, z4, z4, 0);
#endif
}
#endif

// ---------------------------------------------------------------------------
// Weight conversion kernels (f32 -> bf16; transpose produces [N,K] layout)
// ---------------------------------------------------------------------------
__global__ __launch_bounds__(256) void k_convert_bf16(const float* __restrict__ src,
                                                      __bf16* __restrict__ dst, int n) {
  int i = blockIdx.x * 256 + threadIdx.x;
  if (i < n) dst[i] = (__bf16)src[i];
}

__global__ __launch_bounds__(256) void k_transpose_bf16(const float* __restrict__ src,
                                                        __bf16* __restrict__ dst,
                                                        int rows, int cols) {
  int i = blockIdx.x * 256 + threadIdx.x;
  if (i < rows * cols) {
    int r = i / cols, c = i % cols;
    dst[(size_t)c * rows + r] = (__bf16)src[i];
  }
}

// ---------------------------------------------------------------------------
// Tile staging: issue TDM DMA (bf16 operands) or manual loads (f32 A / no TDM)
// ---------------------------------------------------------------------------
template <bool AF32, typename AT>
__device__ __forceinline__ void stage_tiles(const AT* __restrict__ A,
                                            const __bf16* __restrict__ Bw,
                                            __bf16 (*Asb)[32], __bf16 (*Bsb)[32],
                                            int m0, int n0, int kk,
                                            int M, int N, int K, int tid) {
#if USE_TDM
  if (tid < 32) {                 // wave-uniform: only wave 0 issues DMA
    if constexpr (!AF32)
      tdm_load_tile_128x32((unsigned)(size_t)&Asb[0][0],
                           &A[(size_t)m0 * K + kk], (unsigned)K, (unsigned)M);
    tdm_load_tile_128x32((unsigned)(size_t)&Bsb[0][0],
                         &Bw[(size_t)n0 * K + kk], (unsigned)K, (unsigned)N);
  }
#else
  {   // manual B-tile staging (raw bf16 copy, K-contiguous rows)
    const int r   = tid >> 1;     // 0..127 (n)
    const int seg = tid & 1;      // 2 segs x 16 bf16
    const __bf16* bp = &Bw[(size_t)(n0 + r) * K + kk + seg * 16];
    #pragma unroll
    for (int e = 0; e < 16; e++) Bsb[r][seg * 16 + e] = bp[e];
  }
  if constexpr (!AF32) {          // manual bf16 A-tile staging
    const int r   = tid >> 1;
    const int seg = tid & 1;
    const __bf16* ap = &A[(size_t)(m0 + r) * K + kk + seg * 16];
    #pragma unroll
    for (int e = 0; e < 16; e++) Asb[r][seg * 16 + e] = ap[e];
  }
#endif
  if constexpr (AF32) {           // f32 A tile: load + convert + stage
    const int seg   = tid & 7;    // 8 segs x 4 f32
    const int rbase = tid >> 3;   // 0..31
    #pragma unroll
    for (int p = 0; p < 4; p++) {
      const int r = p * 32 + rbase;
      const float* ap = &A[(size_t)(m0 + r) * K + kk + seg * 4];
      #pragma unroll
      for (int e = 0; e < 4; e++) Asb[r][seg * 4 + e] = (__bf16)ap[e];
    }
  }
}

__device__ __forceinline__ void tdm_fence(int tid) {
#if USE_TDM
  if (tid < 32) __builtin_amdgcn_s_wait_tensorcnt(0);
#endif
}

// ---------------------------------------------------------------------------
// bf16 WMMA GEMM:  Cout[M,N] = (relu?)(A[M,K] @ Bw[N,K]^T + bias)
// Double-buffered LDS: TDM/loads for tile kk+32 overlap WMMA on tile kk.
// Block: 256 threads = 8 waves, tile 128x128; wave tile 32x64 (2x4 frags).
// ---------------------------------------------------------------------------
template <typename AT, typename OT, bool BIAS, bool RELU>
__global__ __launch_bounds__(256) void k_gemm(const AT* __restrict__ A,
                                              const __bf16* __restrict__ Bw, // [N,K]
                                              const float* __restrict__ bias,
                                              OT* __restrict__ Cout,
                                              int M, int N, int K) {
  constexpr bool AF32 = is_same_t<AT, float>::value;

  __shared__ alignas(32) __bf16 As[2][128][32];   // As[buf][m][k]
  __shared__ alignas(32) __bf16 Bs[2][128][32];   // Bs[buf][n][k]

  const int tid   = threadIdx.x;
  const int lane  = tid & 31;
  const int wid   = tid >> 5;
  const int wm    = wid & 3;        // 4 x 32 = 128 M
  const int wn    = wid >> 2;       // 2 x 64 = 128 N
  const int m0    = blockIdx.x * 128;
  const int n0    = blockIdx.y * 128;
  const int mhalf = lane >> 4;
  const int ml    = lane & 15;

  v8f acc[2][4] = {};

  // prologue: stage first tile into buffer 0
  stage_tiles<AF32>(A, Bw, As[0], Bs[0], m0, n0, 0, M, N, K, tid);
  tdm_fence(tid);
  __syncthreads();

  int buf = 0;
  for (int kk = 0; kk < K; kk += 32) {
    // issue DMA / loads for the NEXT tile into the inactive buffer
    if (kk + 32 < K)
      stage_tiles<AF32>(A, Bw, As[buf ^ 1], Bs[buf ^ 1], m0, n0, kk + 32,
                        M, N, K, tid);

    // --- fragment loads, contiguous k = 16*mhalf + s for A and B (current buf) ---
    v16bf afrag[2], bfrag[4];
    #pragma unroll
    for (int i = 0; i < 2; i++) {
      const __bf16* arow = &As[buf][wm * 32 + i * 16 + ml][0];
      const int h8 = mhalf * 16;
      #pragma unroll
      for (int j = 0; j < 8; j++) {
        afrag[i][j]     = arow[h8 + j];        // K = h8..h8+7
        afrag[i][8 + j] = arow[h8 + 8 + j];    // K = h8+8..h8+15
      }
    }
    #pragma unroll
    for (int j = 0; j < 4; j++) {
      const __bf16* brow = &Bs[buf][wn * 64 + j * 16 + ml][mhalf * 16];
      bfrag[j] = *(const v16bf*)brow;          // 16 contiguous bf16 (32B)
    }

    #pragma unroll
    for (int i = 0; i < 2; i++)
      #pragma unroll
      for (int j = 0; j < 4; j++)
        acc[i][j] = __builtin_amdgcn_wmma_f32_16x16x32_bf16(
            false, afrag[i], false, bfrag[j], (short)0, acc[i][j], false, false);

    // next tile must be complete and all reads of current buf retired
    tdm_fence(tid);
    __syncthreads();
    buf ^= 1;
  }

  // --- epilogue: bias / relu / store (f32 or bf16) ---
  #pragma unroll
  for (int i = 0; i < 2; i++) {
    #pragma unroll
    for (int j = 0; j < 4; j++) {
      const int nn = n0 + wn * 64 + j * 16 + ml;
      float bv = 0.f;
      if constexpr (BIAS) bv = bias[nn];
      #pragma unroll
      for (int r = 0; r < 8; r++) {
        const int mm = m0 + wm * 32 + i * 16 + mhalf * 8 + r;
        float v = acc[i][j][r];
        if constexpr (BIAS) v += bv;
        if constexpr (RELU) v = v > 0.f ? v : 0.f;
        Cout[(size_t)mm * N + nn] = (OT)v;
      }
    }
  }
}

// ---------------------------------------------------------------------------
// Chunked parallel scan:  h_t = a*h_{t-1} + u_t   (U is f32, Hs emitted bf16)
// ---------------------------------------------------------------------------
__global__ __launch_bounds__(256) void k_scan_local(float* __restrict__ U,
                                                    const float* __restrict__ a,
                                                    float* __restrict__ carry) {
  const int h = threadIdx.x;
  const int b = blockIdx.x / NC;
  const int c = blockIdx.x % NC;
  const float av = a[h];
  size_t base = ((size_t)b * TLEN + (size_t)c * LCHUNK) * HDIM + h;
  float hloc = 0.f;
  for (int k = 0; k < LCHUNK; k++) {
    float u = U[base + (size_t)k * HDIM];
    hloc = av * hloc + u;
    U[base + (size_t)k * HDIM] = hloc;
  }
  carry[((size_t)b * NC + c) * HDIM + h] = hloc;
}

__global__ __launch_bounds__(256) void k_scan_carry(const float* __restrict__ carry,
                                                    const float* __restrict__ a,
                                                    float* __restrict__ hin) {
  const int h = threadIdx.x;
  const int b = blockIdx.x;
  const float av = a[h];
  float aL = av;
  #pragma unroll
  for (int i = 0; i < 8; i++) aL = aL * aL;   // av^256 = av^LCHUNK
  float Hin = 0.f;
  for (int c = 0; c < NC; c++) {
    size_t idx = ((size_t)b * NC + c) * HDIM + h;
    hin[idx] = Hin;                           // incoming state for chunk c
    Hin = aL * Hin + carry[idx];
  }
}

__global__ __launch_bounds__(256) void k_scan_fixup(const float* __restrict__ U,
                                                    const float* __restrict__ a,
                                                    const float* __restrict__ hin,
                                                    __bf16* __restrict__ Hs) {
  const int h = threadIdx.x;
  const int b = blockIdx.x / NC;
  const int c = blockIdx.x % NC;
  const float av = a[h];
  const float Hin = hin[((size_t)b * NC + c) * HDIM + h];  // 0 for c==0
  size_t base = ((size_t)b * TLEN + (size_t)c * LCHUNK) * HDIM + h;
  float ap = av;                              // a^{k+1}
  for (int k = 0; k < LCHUNK; k++) {
    float hv = U[base + (size_t)k * HDIM] + ap * Hin;
    Hs[base + (size_t)k * HDIM] = (__bf16)hv;
    ap *= av;
  }
}

// ---------------------------------------------------------------------------
extern "C" void kernel_launch(void* const* d_in, const int* in_sizes, int n_in,
                              void* d_out, int out_size, void* d_ws, size_t ws_size,
                              hipStream_t stream) {
  const float* x  = (const float*)d_in[0];   // [8,4096,512]
  const float* a  = (const float*)d_in[1];   // [256]
  const float* Bm = (const float*)d_in[2];   // [P,H]
  const float* Cm = (const float*)d_in[3];   // [H,P]
  const float* Wi = (const float*)d_in[4];   // [P,D_IN]  -> already [N,K]
  const float* bi = (const float*)d_in[5];   // [P]
  const float* Wo = (const float*)d_in[6];   // [D_OUT,P] -> already [N,K]
  const float* bo = (const float*)d_in[7];   // [D_OUT]
  float* out = (float*)d_out;                // [8,4096,512]

  char* ws = (char*)d_ws;
  size_t off = 0;
  auto carve = [&](size_t bytes) -> void* {
    void* p = ws + off;
    off += (bytes + 255) & ~(size_t)255;
    return p;
  };
  __bf16* Wn    = (__bf16*)carve((size_t)PDIM * D_IN * 2);    // [256,512] = [N,K]
  __bf16* Bt    = (__bf16*)carve((size_t)HDIM * PDIM * 2);    // [H,P]     = [N,K]
  __bf16* Ct    = (__bf16*)carve((size_t)PDIM * HDIM * 2);    // [P,H]     = [N,K]
  __bf16* Won   = (__bf16*)carve((size_t)D_OUT * PDIM * 2);   // [512,256] = [N,K]
  float*  carry = (float*)carve((size_t)BSZ * NC * HDIM * 4);
  float*  hin   = (float*)carve((size_t)BSZ * NC * HDIM * 4);
  __bf16* Xp    = (__bf16*)carve((size_t)MROWS * PDIM * 2);   // 16 MB (reused as Y)
  float*  U     = (float*)carve((size_t)MROWS * HDIM * 4);    // 32 MB (scan in place)
  __bf16* Hs    = (__bf16*)carve((size_t)MROWS * HDIM * 2);   // 16 MB

  // weights -> bf16 in [N,K]; Wi/Wo are already [N,K], B/C need transpose
  k_convert_bf16 <<<(PDIM * D_IN + 255) / 256, 256, 0, stream>>>(Wi, Wn, PDIM * D_IN);
  k_transpose_bf16<<<(PDIM * HDIM + 255) / 256, 256, 0, stream>>>(Bm, Bt, PDIM, HDIM);
  k_transpose_bf16<<<(HDIM * PDIM + 255) / 256, 256, 0, stream>>>(Cm, Ct, HDIM, PDIM);
  k_convert_bf16 <<<(D_OUT * PDIM + 255) / 256, 256, 0, stream>>>(Wo, Won, D_OUT * PDIM);

  // 1) Xp = relu(X @ Wi^T + bi)  -> bf16   [32768,256], K=512
  k_gemm<float, __bf16, true, true><<<dim3(MROWS / 128, PDIM / 128), 256, 0, stream>>>(
      x, Wn, bi, Xp, MROWS, PDIM, D_IN);
  // 2) U = Xp @ B                -> f32    [32768,256], K=256
  k_gemm<__bf16, float, false, false><<<dim3(MROWS / 128, HDIM / 128), 256, 0, stream>>>(
      Xp, Bt, nullptr, U, MROWS, HDIM, PDIM);
  // 3) chunked scan: U (in place) -> Hs (bf16)
  k_scan_local<<<BSZ * NC, 256, 0, stream>>>(U, a, carry);
  k_scan_carry<<<BSZ, 256, 0, stream>>>(carry, a, hin);
  k_scan_fixup<<<BSZ * NC, 256, 0, stream>>>(U, a, hin, Hs);
  // 4) Y = Hs @ C                -> bf16 (reuse Xp)  [32768,256], K=256
  k_gemm<__bf16, __bf16, false, false><<<dim3(MROWS / 128, PDIM / 128), 256, 0, stream>>>(
      Hs, Ct, nullptr, Xp, MROWS, PDIM, HDIM);
  // 5) out = Y @ Wo^T + bo       -> f32   [32768,512], K=256
  k_gemm<__bf16, float, true, false><<<dim3(MROWS / 128, D_OUT / 128), 256, 0, stream>>>(
      Xp, Won, bo, out, MROWS, D_OUT, PDIM);
}
